// RNN_9560597201285
// MI455X (gfx1250) — matrix operand order repaired and checked by reading, not compiled, not run on hardware
//
#include <hip/hip_runtime.h>

// ---------------------------------------------------------------------------
// RNN forward for MI455X (gfx1250), wave32 + WMMA f32_16x16x32_f16.
//
//   xproj = input @ wi + brec          (GEMM, WMMA, memory-bound)
//   h_t   = 0.5*h + 0.5*tanh(h@wrec^T + xproj_t)   (persistent 1-WG scan)
//   out   = hs @ wo                    (GEMM, WMMA, memory-bound)
//
// Scan kernel: h (f16) + 192 K-columns of wrec (f16) resident in LDS (bank-
// conflict-free padded strides), wrec slab staged once via
// global_load_async_to_lds_b128 (ASYNCcnt), rest of wrec streams from L2.
//
// Workspace layout (bytes):
//   [0,            134217728) : xproj  f32  [1024][64][512]   ([t][b][h])
//   [134217728,    201326592) : hs16   f16  [64][1024][512]   ([b][t][h])
//   [201326592,    201392128) : wiT    f16  [512][64]
//   [201392128,    201916416) : wrec16 f16  [512][512]  (row n, k contiguous)
//   [201916416,    201981952) : woT    f16  [64][512]
// ---------------------------------------------------------------------------

typedef _Float16 h8_t  __attribute__((ext_vector_type(8)));
typedef _Float16 h16_t __attribute__((ext_vector_type(16)));
typedef float    f8_t  __attribute__((ext_vector_type(8)));

#define DT_STEP 0.5f

#define HS_STRIDE 520  // 64x520 f16 h buffer: 260 dwords/row == 4 mod 64 banks
#define WL_KCOLS  192  // K columns of wrec cached in LDS (6 of 16 k-steps)
#define WL_STRIDE 200  // 100 dwords/row == 36 mod 64 banks -> conflict-free

// ---- WMMA fragment loaders (layouts per CDNA5 ISA §7.12.2) ------

// A fragment: 16x32 f16, row-major source (ldm = row stride in elements).
// lanes 0-15: row M=lane,    K = {kb..kb+7, kb+16..kb+23}
// lanes 16-31: row M=lane-16, K = {kb+8..kb+15, kb+24..kb+31}
__device__ __forceinline__ h16_t load_a_frag(const _Float16* base, int ldm,
                                             int kbase, int lane) {
  int row = lane & 15;
  int hi  = lane >> 4;
  const _Float16* p = base + row * ldm + kbase + hi * 8;
  h8_t lo = *(const h8_t*)(p);
  h8_t up = *(const h8_t*)(p + 16);
  return __builtin_shufflevector(lo, up, 0, 1, 2, 3, 4, 5, 6, 7,
                                 8, 9, 10, 11, 12, 13, 14, 15);
}

// Same, but converts from an f32 row-major source on the fly.
__device__ __forceinline__ h16_t load_a_frag_f32(const float* base, int ldm,
                                                 int kbase, int lane) {
  int row = lane & 15;
  int hi  = lane >> 4;
  const float* p = base + row * ldm + kbase + hi * 8;
  h16_t r;
#pragma unroll
  for (int i = 0; i < 8; ++i) {
    r[i]     = (_Float16)p[i];
    r[8 + i] = (_Float16)p[16 + i];
  }
  return r;
}

// B fragment: 32x16 f16, loaded from B^T stored row-major [N][K].
// lanes 0-15: col N=lane, K = {kb..kb+7, kb+16..kb+23} ; lanes 16-31: K = {kb+8..kb+15, kb+24..kb+31}.
__device__ __forceinline__ h16_t load_b_frag(const _Float16* baseT, int ldm,
                                             int kbase, int lane) {
  int n  = lane & 15;
  int hi = lane >> 4;
  const _Float16* p = baseT + n * ldm + kbase + hi * 8;
  h8_t lo = *(const h8_t*)(p);
  h8_t up = *(const h8_t*)(p + 16);
  return __builtin_shufflevector(lo, up, 0, 1, 2, 3, 4, 5, 6, 7,
                                 8, 9, 10, 11, 12, 13, 14, 15);
}

__device__ __forceinline__ f8_t wmma_f16(h16_t a, h16_t b, f8_t c) {
  return __builtin_amdgcn_wmma_f32_16x16x32_f16(false, a, false, b,
                                                (short)0, c, false, false);
}

__device__ __forceinline__ float fast_tanh(float x) {
  x = fminf(fmaxf(x, -15.0f), 15.0f);
  float e = __expf(2.0f * x);
  return (e - 1.0f) / (e + 1.0f);
}

// ---- prep: f32 -> f16 weight conversion / transposition --------------------
__global__ void __launch_bounds__(256)
prep_weights(const float* __restrict__ wi, const float* __restrict__ wrec,
             const float* __restrict__ wo, _Float16* __restrict__ wiT,
             _Float16* __restrict__ wrec16, _Float16* __restrict__ woT) {
  int idx = blockIdx.x * blockDim.x + threadIdx.x;  // grid covers 512*512
  if (idx < 512 * 512) wrec16[idx] = (_Float16)wrec[idx];  // B^T == wrec
  if (idx < 64 * 512) {
    int n = idx >> 6, k = idx & 63;     // wiT[n][k] = wi[k][n]
    wiT[idx] = (_Float16)wi[k * 512 + n];
    int o = idx >> 9, kk = idx & 511;   // woT[o][k] = wo[k][o]
    woT[idx] = (_Float16)wo[kk * 64 + o];
  }
}

// ---- kernel 1: xproj = input @ wi + brec  ([65536x64]·[64x512]) ------------
// Output layout [t][b][h] so the scan consumes one contiguous 128KB slab/step.
__global__ void __launch_bounds__(128)
xproj_kernel(const float* __restrict__ input, const _Float16* __restrict__ wiT,
             const float* __restrict__ brec, float* __restrict__ xproj) {
  int lane = threadIdx.x & 31;
  int wave = threadIdx.x >> 5;
  int tile = blockIdx.x * 4 + wave;
  int mt = tile >> 5;  // row tile over B*T (0..4095), row = b*1024 + t
  int nt = tile & 31;  // col tile over H   (0..31)

  const float*    abase = input + mt * 16 * 64;
  const _Float16* bbase = wiT + nt * 16 * 64;

  f8_t acc;
#pragma unroll
  for (int r = 0; r < 8; ++r) acc[r] = 0.0f;

#pragma unroll
  for (int ks = 0; ks < 2; ++ks) {
    h16_t a = load_a_frag_f32(abase, 64, ks * 32, lane);
    h16_t b = load_b_frag(bbase, 64, ks * 32, lane);
    acc = wmma_f16(a, b, acc);
  }

  int n    = nt * 16 + (lane & 15);
  int hi   = lane >> 4;
  float bv = brec[n];
#pragma unroll
  for (int r = 0; r < 8; ++r) {
    int row = mt * 16 + r + hi * 8;  // = b*1024 + t
    int b   = row >> 10;
    int t   = row & 1023;
    xproj[(t * 64 + b) * 512 + n] = acc[r] + bv;  // [t][b][h]
  }
}

// ---- kernel 2: persistent recurrent scan (single workgroup, 16 waves) ------
__global__ void __launch_bounds__(512)
recur_kernel(const float* __restrict__ xproj, const float* __restrict__ h0,
             const _Float16* __restrict__ wrec16, _Float16* __restrict__ hs16) {
  // 64*520*2 = 66,560 B  +  512*200*2 = 204,800 B  => 271,360 B < 320 KB/WGP
  __shared__ _Float16 hsh[64 * HS_STRIDE];   // h state, f16, padded stride
  __shared__ _Float16 wlds[512 * WL_STRIDE]; // wrec[:, 0:192], padded stride

  int tid  = threadIdx.x;
  int lane = tid & 31;
  int wave = tid >> 5;
  int hi   = lane >> 4;
  int nloc = lane & 15;
  int mt     = wave & 3;         // batch tile (0..3)
  int ntBase = (wave >> 2) * 8;  // 8 H-tiles per wave (0,8,16,24)

  // ---- stage wrec[:, 0:WL_KCOLS] into LDS with async b128 copies ----------
  // 512 rows x 192 halves = 12288 16B chunks; 24 chunks per thread.
#pragma unroll 4
  for (int i = 0; i < 24; ++i) {
    int c  = i * 512 + tid;       // chunk id
    int n  = c / 24;              // wrec row
    int kc = (c % 24) * 8;        // half offset within row (0..184)
    unsigned int      laddr = (unsigned int)(unsigned long long)
                              (&wlds[n * WL_STRIDE + kc]);
    unsigned long long gaddr = (unsigned long long)
                               (const void*)(wrec16 + n * 512 + kc);
    asm volatile("global_load_async_to_lds_b128 %0, %1, off"
                 :: "v"(laddr), "v"(gaddr) : "memory");
  }

  // broadcast h0 over batch into LDS (padded stride)
  for (int idx = tid; idx < 64 * 512; idx += 512) {
    int b = idx >> 9, n = idx & 511;
    hsh[b * HS_STRIDE + n] = (_Float16)h0[n];
  }

  // f32 master copy of h in registers, WMMA C/D layout
  f8_t hreg[8];
#pragma unroll
  for (int j = 0; j < 8; ++j) {
    float v = h0[(ntBase + j) * 16 + nloc];
#pragma unroll
    for (int r = 0; r < 8; ++r) hreg[j][r] = v;
  }

  asm volatile("s_wait_asynccnt 0" ::: "memory");  // wrec slab landed in LDS
  __syncthreads();

  const _Float16* ash = &hsh[mt * 16 * HS_STRIDE];

  for (int t = 0; t < 1024; ++t) {
    f8_t acc[8];
#pragma unroll
    for (int j = 0; j < 8; ++j)
#pragma unroll
      for (int r = 0; r < 8; ++r) acc[j][r] = 0.0f;

    // k-steps 0..5: B operand from LDS-resident wrec slab (conflict-free)
#pragma unroll 2
    for (int ks = 0; ks < 6; ++ks) {
      h16_t a = load_a_frag(ash, HS_STRIDE, ks * 32, lane);
#pragma unroll
      for (int j = 0; j < 8; ++j) {
        h16_t b = load_b_frag(&wlds[(ntBase + j) * 16 * WL_STRIDE], WL_STRIDE,
                              ks * 32, lane);
        acc[j] = wmma_f16(a, b, acc[j]);
      }
    }
    // k-steps 6..15: B operand streams from L2-resident wrec
#pragma unroll 2
    for (int ks = 6; ks < 16; ++ks) {
      h16_t a = load_a_frag(ash, HS_STRIDE, ks * 32, lane);
#pragma unroll
      for (int j = 0; j < 8; ++j) {
        h16_t b =
            load_b_frag(wrec16 + (ntBase + j) * 16 * 512, 512, ks * 32, lane);
        acc[j] = wmma_f16(a, b, acc[j]);
      }
    }

    // h_new = (1-dt)*h + dt*tanh(acc + xproj[t]); write hs (f16) to global
    const float* xslab = xproj + t * 64 * 512;  // contiguous 128KB per step
#pragma unroll
    for (int j = 0; j < 8; ++j) {
      int n = (ntBase + j) * 16 + nloc;
#pragma unroll
      for (int r = 0; r < 8; ++r) {
        int b    = mt * 16 + r + hi * 8;  // batch index 0..63
        float x  = xslab[b * 512 + n];
        float v  = fast_tanh(acc[j][r] + x);
        float hn = (1.0f - DT_STEP) * hreg[j][r] + DT_STEP * v;
        hreg[j][r] = hn;
        hs16[(b * 1024 + t) * 512 + n] = (_Float16)hn;
      }
      if (t + 1 < 1024) {  // prefetch next step's xproj slab into caches
        int b0 = mt * 16 + hi * 8;
        __builtin_prefetch(&xproj[((t + 1) * 64 + b0) * 512 + n], 0, 1);
      }
    }

    __syncthreads();  // everyone done reading hsh for step t
#pragma unroll
    for (int j = 0; j < 8; ++j) {
      int n = (ntBase + j) * 16 + nloc;
#pragma unroll
      for (int r = 0; r < 8; ++r) {
        int b = mt * 16 + r + hi * 8;
        hsh[b * HS_STRIDE + n] = (_Float16)hreg[j][r];
      }
    }
    __syncthreads();  // hsh ready for step t+1
  }
}

// ---- kernel 3: out = hs @ wo  ([65536x512]·[512x64]) -----------------------
__global__ void __launch_bounds__(256)
out_kernel(const _Float16* __restrict__ hs16, const _Float16* __restrict__ woT,
           float* __restrict__ out) {
  int lane = threadIdx.x & 31;
  int wave = threadIdx.x >> 5;
  int tile = blockIdx.x * 8 + wave;  // 16384 tiles
  int mt = tile >> 2;                // row tile over B*T
  int ot = tile & 3;                 // col tile over O

  const _Float16* abase = hs16 + mt * 16 * 512;
  const _Float16* bbase = woT + ot * 16 * 512;

  f8_t acc;
#pragma unroll
  for (int r = 0; r < 8; ++r) acc[r] = 0.0f;

#pragma unroll 4
  for (int ks = 0; ks < 16; ++ks) {
    h16_t a = load_a_frag(abase, 512, ks * 32, lane);
    h16_t b = load_b_frag(bbase, 512, ks * 32, lane);
    acc = wmma_f16(a, b, acc);
  }

  int n  = ot * 16 + (lane & 15);
  int hi = lane >> 4;
#pragma unroll
  for (int r = 0; r < 8; ++r)
    out[(mt * 16 + r + hi * 8) * 64 + n] = acc[r];
}

// ---------------------------------------------------------------------------
extern "C" void kernel_launch(void* const* d_in, const int* in_sizes, int n_in,
                              void* d_out, int out_size, void* d_ws,
                              size_t ws_size, hipStream_t stream) {
  const float* input = (const float*)d_in[0];  // [64,1024,64]
  const float* wi    = (const float*)d_in[1];  // [64,512]
  const float* wrec  = (const float*)d_in[2];  // [512,512]
  const float* wo    = (const float*)d_in[3];  // [512,64]
  const float* brec  = (const float*)d_in[4];  // [512]
  const float* h0    = (const float*)d_in[5];  // [512]
  float* out = (float*)d_out;                  // [64,1024,64]

  char* ws = (char*)d_ws;
  float*    xproj  = (float*)ws;                       // 134,217,728 B
  _Float16* hs16   = (_Float16*)(ws + 134217728);      //  67,108,864 B
  _Float16* wiT    = (_Float16*)(ws + 201326592);      //      65,536 B
  _Float16* wrec16 = (_Float16*)(ws + 201392128);      //     524,288 B
  _Float16* woT    = (_Float16*)(ws + 201916416);      //      65,536 B

  prep_weights<<<1024, 256, 0, stream>>>(wi, wrec, wo, wiT, wrec16, woT);
  xproj_kernel<<<32768, 128, 0, stream>>>(input, wiT, brec, xproj);
  recur_kernel<<<1, 512, 0, stream>>>(xproj, h0, wrec16, hs16);
  out_kernel<<<2048, 256, 0, stream>>>(hs16, woT, out);
}
